// GroupQuerySelfAttention_14534169329932
// MI455X (gfx1250) — matrix operand order repaired and checked by reading, not compiled, not run on hardware
//
#include <hip/hip_runtime.h>
#include <hip/hip_bf16.h>

// ---------------------------------------------------------------------------
// GQA self-attention for gfx1250 (MI455X).
// All matmuls via v_wmma_f32_16x16x32_bf16; projection GEMMs use async
// global->LDS staging (ASYNCcnt) with double buffering; attention is a
// flash-style streaming kernel with ds_swizzle softmax reductions.
// B=2, S=2048, D_MODEL=1024, N_HEADS=16, D_HEAD=64, N_KV_GROUPS=4, KV_DIM=256.
// ---------------------------------------------------------------------------

#define BB 2
#define SS 2048
#define DM 1024
#define NH 16
#define DH 64
#define NG 4
#define KVD 256

typedef __attribute__((ext_vector_type(16))) __bf16 v16bf;
typedef __attribute__((ext_vector_type(8)))  __bf16 v8bf;
typedef __attribute__((ext_vector_type(8)))  float  v8f;
typedef __attribute__((ext_vector_type(4)))  int    v4i;

// ---- bf16 helpers (round-to-nearest-even) ---------------------------------
__device__ __forceinline__ unsigned short bf16_bits(float f) {
    union { float f; unsigned u; } v; v.f = f;
    return (unsigned short)((v.u + 0x7FFFu + ((v.u >> 16) & 1u)) >> 16);
}
__device__ __forceinline__ __bf16 f2bf(float f) {
    unsigned short b = bf16_bits(f);
    __bf16 r; __builtin_memcpy(&r, &b, 2); return r;
}

// ---- WMMA fragment loaders -------------------------------------------------
// A-matrix 16x32 bf16 (ISA layout): lane L row M=L%16; h=L/16:
//   e0..7 -> K = k0 + h*8 + i ; e8..15 -> K = k0 + 16 + h*8 + i
// caller passes p = &A[row*ld + k0 + h*8]
__device__ __forceinline__ v16bf ld_a16(const __bf16* p) {
    v8bf lo = *(const v8bf*)p;
    v8bf hi = *(const v8bf*)(p + 16);
    return __builtin_shufflevector(lo, hi, 0,1,2,3,4,5,6,7,8,9,10,11,12,13,14,15);
}
// B-matrix 32x16 bf16: lane L column N=L%16; h=L/16:
//   e0..7 -> K = k0 + h*8 + i ; e8..15 -> K = k0 + 16 + h*8 + i
// caller passes p = &BT[n*ld + k0 + h*8]
__device__ __forceinline__ v16bf ld_b16(const __bf16* p) {
    v8bf lo = *(const v8bf*)p;
    v8bf hi = *(const v8bf*)(p + 16);
    return __builtin_shufflevector(lo, hi, 0,1,2,3,4,5,6,7,8,9,10,11,12,13,14,15);
}

__device__ __forceinline__ v8f wmma_bf16(v16bf a, v16bf b, v8f c) {
    return __builtin_amdgcn_wmma_f32_16x16x32_bf16(false, a, false, b, (short)0, c, false, false);
}

// ---- async global->LDS copy (CDNA5 ASYNCcnt path), 16B granularity --------
typedef __attribute__((address_space(1))) v4i* g_v4i_p;   // global int4*
typedef __attribute__((address_space(3))) v4i* l_v4i_p;   // LDS int4*

__device__ __forceinline__ void cp_async16(void* l, const void* g) {
#if defined(__has_builtin) && __has_builtin(__builtin_amdgcn_global_load_async_to_lds_b128)
    __builtin_amdgcn_global_load_async_to_lds_b128(
        (g_v4i_p)(unsigned long long)g,
        (l_v4i_p)(unsigned)(unsigned long long)l,
        0, 0);
#else
    *(v8bf*)l = *(const v8bf*)g;   // fallback: vector load + ds_store_b128
#endif
}
__device__ __forceinline__ void wait_async0() {
#if defined(__has_builtin) && __has_builtin(__builtin_amdgcn_s_wait_asynccnt)
    __builtin_amdgcn_s_wait_asynccnt(0);
#else
    asm volatile("s_wait_asynccnt 0" ::: "memory");
#endif
}

// ---- 16-lane row reductions via immediate ds_swizzle (xor < 16 keeps groups)
#define SWZF(v, imm) __int_as_float(__builtin_amdgcn_ds_swizzle(__float_as_int(v), (imm)))
__device__ __forceinline__ float redmax16(float v) {
    v = fmaxf(v, SWZF(v, 0x201f));   // xor 8
    v = fmaxf(v, SWZF(v, 0x101f));   // xor 4
    v = fmaxf(v, SWZF(v, 0x081f));   // xor 2
    v = fmaxf(v, SWZF(v, 0x041f));   // xor 1
    return v;
}
__device__ __forceinline__ float redsum16(float v) {
    v += SWZF(v, 0x201f);
    v += SWZF(v, 0x101f);
    v += SWZF(v, 0x081f);
    v += SWZF(v, 0x041f);
    return v;
}

// ===========================================================================
// Kernel 1: f32 -> bf16 flat convert, 4 elements/thread
// ===========================================================================
__global__ __launch_bounds__(256) void k_cvt4(const float4* __restrict__ src,
                                              ushort4* __restrict__ dst, int n4) {
    int i = blockIdx.x * 256 + threadIdx.x;
    if (i < n4) {
        float4 f = src[i];
        ushort4 u;
        u.x = bf16_bits(f.x); u.y = bf16_bits(f.y);
        u.z = bf16_bits(f.z); u.w = bf16_bits(f.w);
        dst[i] = u;
    }
}

// ===========================================================================
// Kernel 2: weight transpose + convert: W[K][N] f32 -> WT[N][K] bf16
// ===========================================================================
__global__ __launch_bounds__(256) void k_wtrans(const float* __restrict__ src,
                                                __bf16* __restrict__ dst, int K, int N) {
    int i = blockIdx.x * 256 + threadIdx.x;
    if (i < K * N) {
        int kk = i / N, nn = i - kk * N;
        dst[(size_t)nn * K + kk] = f2bf(src[i]);
    }
}

// ===========================================================================
// Kernel 3: block-tiled GEMM  out[M][N] = A[M][K] * WT[N][K]^T + bias
// Block (8 waves) computes 128x64; per 32-wide K step the block stages
// A(128x32) + WT(64x32) bf16 tiles in double-buffered LDS via async copies,
// then each wave runs 4 WMMAs on its 16x64 slice.
// ===========================================================================
template <bool OUT_BF16>
__global__ __launch_bounds__(256) void k_gemm(const __bf16* __restrict__ A,
                                              const __bf16* __restrict__ BT,
                                              const float* __restrict__ bias,
                                              void* __restrict__ out,
                                              int M, int N, int K) {
    __shared__ __bf16 sA[2][128 * 32];   // 8 KB x2
    __shared__ __bf16 sB[2][64 * 32];    // 4 KB x2

    int t    = threadIdx.x;
    int w    = t >> 5;
    int lane = t & 31;
    int l16  = lane & 15, hf = lane >> 4;

    int nb  = N >> 6;                     // block-cols
    int bm0 = (blockIdx.x / nb) << 7;     // 128-row tile base
    int bn0 = (blockIdx.x % nb) << 6;     // 64-col tile base

    // stage one K-slab (k0..k0+31) into buffer `buf`
    auto stage = [&](int buf, int k0) {
#pragma unroll
        for (int i = 0; i < 2; ++i) {     // A: 512 chunks of 16B over 256 thr
            int c = t + 256 * i;
            int row = c >> 2, seg = c & 3;
            cp_async16(&sA[buf][row * 32 + seg * 8],
                       A + (size_t)(bm0 + row) * K + k0 + seg * 8);
        }
        int n = t >> 2, seg = t & 3;      // B: 256 chunks of 16B
        cp_async16(&sB[buf][n * 32 + seg * 8],
                   BT + (size_t)(bn0 + n) * K + k0 + seg * 8);
    };

    v8f acc[4] = {{}, {}, {}, {}};
    stage(0, 0);
    int cur = 0;
    for (int k0 = 0; k0 < K; k0 += 32) {
        wait_async0();
        __syncthreads();
        if (k0 + 32 < K) stage(cur ^ 1, k0 + 32);

        v16bf af = ld_a16(&sA[cur][(w * 16 + l16) * 32 + hf * 8]);
        const __bf16* bb = &sB[cur][l16 * 32 + hf * 8];
        acc[0] = wmma_bf16(af, ld_b16(bb),           acc[0]);
        acc[1] = wmma_bf16(af, ld_b16(bb + 16 * 32), acc[1]);
        acc[2] = wmma_bf16(af, ld_b16(bb + 32 * 32), acc[2]);
        acc[3] = wmma_bf16(af, ld_b16(bb + 48 * 32), acc[3]);
        cur ^= 1;
    }

    int m0 = bm0 + w * 16 + hf * 8;
#pragma unroll
    for (int nt = 0; nt < 4; ++nt) {
        int n = bn0 + nt * 16 + l16;
        float bv = bias[n];
#pragma unroll
        for (int r = 0; r < 8; ++r) {
            float v = acc[nt][r] + bv;
            if (OUT_BF16) ((__bf16*)out)[(size_t)(m0 + r) * N + n] = f2bf(v);
            else          ((float*) out)[(size_t)(m0 + r) * N + n] = v;
        }
    }
}

// ===========================================================================
// Kernel 4: V[b][s][g*64+d] bf16 -> VT[b][g][d][s] bf16
// ===========================================================================
__global__ __launch_bounds__(256) void k_vtrans(const __bf16* __restrict__ src,
                                                __bf16* __restrict__ dst) {
    int i = blockIdx.x * 256 + threadIdx.x;           // over B*S*KVD
    int b = i / (SS * KVD);
    int rem = i - b * (SS * KVD);
    int s = rem / KVD;
    int c = rem - s * KVD;
    int g = c >> 6, d = c & 63;
    dst[(((size_t)(b * NG + g) * DH + d) * SS) + s] = src[i];
}

// ===========================================================================
// Kernel 5: flash attention. One wave = one 16-row Q tile of one head.
// 32 keys/iter: 4 WMMAs QK^T, online softmax (ds_swizzle reductions),
// LDS C->A relayout of P, 4 WMMAs P*V into 16x64 f32 accumulator.
// ===========================================================================
__global__ __launch_bounds__(256) void k_attn(const __bf16* __restrict__ Qb,
                                              const __bf16* __restrict__ Kb,
                                              const __bf16* __restrict__ Vtb,
                                              __bf16* __restrict__ ctx) {
    __shared__ __bf16 plds[8 * 16 * 32];              // 8 waves * 16x32 P tile
    int warp = threadIdx.x >> 5;
    int lane = threadIdx.x & 31;
    int l16 = lane & 15, hf = lane >> 4;

    int wid = blockIdx.x * 8 + warp;                  // 0 .. B*NH*(S/16)-1
    int qt = wid & 127;
    int h  = (wid >> 7) & 15;
    int b  = wid >> 11;
    int g  = h >> 2;

    const __bf16* Qp = Qb  + ((size_t)(b * SS + qt * 16) * DM + h * DH);
    const __bf16* Kp = Kb  + ((size_t)b * SS * KVD + g * DH);
    const __bf16* Vt = Vtb + ((size_t)(b * NG + g) * DH * SS);
    __bf16* pl = plds + warp * 512;

    // Q A-fragments for d=0..31 and d=32..63 (reused over all key chunks)
    const __bf16* qbase = Qp + (size_t)l16 * DM + hf * 8;
    v16bf qa0 = ld_a16(qbase);
    v16bf qa1 = ld_a16(qbase + 32);

    v8f o0 = {}, o1 = {}, o2 = {}, o3 = {};
    float mrow[8], lrow[8];
#pragma unroll
    for (int r = 0; r < 8; ++r) { mrow[r] = -1e30f; lrow[r] = 0.0f; }

    const float scale = 0.125f;                       // 1/sqrt(64)

    for (int kv = 0; kv < SS; kv += 32) {
        // ---- scores: two 16-key tiles, K-dim = d (64) -> 2 WMMA each ------
        const __bf16* kb0 = Kp + (size_t)(kv + l16) * KVD + hf * 8;
        const __bf16* kb1 = kb0 + (size_t)16 * KVD;
        v8f s0 = {}, s1 = {};
        s0 = wmma_bf16(qa0, ld_b16(kb0),      s0);
        s0 = wmma_bf16(qa1, ld_b16(kb0 + 32), s0);
        s1 = wmma_bf16(qa0, ld_b16(kb1),      s1);
        s1 = wmma_bf16(qa1, ld_b16(kb1 + 32), s1);

        // prefetch next chunk of K and V^T streams into cache
        if (kv + 32 < SS) {
            __builtin_prefetch(Kp + (size_t)(kv + 32 + l16) * KVD, 0, 0);
            __builtin_prefetch(Vt + (size_t)l16 * SS + kv + 32, 0, 0);
        }

        // ---- online softmax over this 32-key chunk ------------------------
        float p0[8], p1[8];
#pragma unroll
        for (int r = 0; r < 8; ++r) {
            float a = s0[r] * scale;
            float c = s1[r] * scale;
            float mx   = redmax16(fmaxf(a, c));
            float mnew = fmaxf(mrow[r], mx);
            float alpha = __expf(mrow[r] - mnew);
            p0[r] = __expf(a - mnew);
            p1[r] = __expf(c - mnew);
            float rs = redsum16(p0[r] + p1[r]);
            lrow[r] = lrow[r] * alpha + rs;
            mrow[r] = mnew;
            o0[r] *= alpha; o1[r] *= alpha; o2[r] *= alpha; o3[r] *= alpha;
        }

        // ---- C-layout -> A-layout relayout of P through (wave-private) LDS
#pragma unroll
        for (int r = 0; r < 8; ++r) {
            pl[(hf * 8 + r) * 32 + l16]      = f2bf(p0[r]);
            pl[(hf * 8 + r) * 32 + 16 + l16] = f2bf(p1[r]);
        }
        asm volatile("s_wait_dscnt 0" ::: "memory");
        v16bf pa = ld_a16(pl + l16 * 32 + hf * 8);

        // ---- context accumulate: P(16x32) * V(32x64) ----------------------
        const __bf16* vb = Vt + (size_t)l16 * SS + kv + hf * 8;
        o0 = wmma_bf16(pa, ld_b16(vb),           o0);
        o1 = wmma_bf16(pa, ld_b16(vb + 16 * SS), o1);
        o2 = wmma_bf16(pa, ld_b16(vb + 32 * SS), o2);
        o3 = wmma_bf16(pa, ld_b16(vb + 48 * SS), o3);
    }

    // ---- normalize (v_rcp_f32) and emit ctx[b][s][h*64 + d] as bf16 -------
#pragma unroll
    for (int r = 0; r < 8; ++r) {
        float inv = __builtin_amdgcn_rcpf(lrow[r]);
        size_t row = (size_t)(b * SS + qt * 16 + hf * 8 + r) * DM + h * DH;
        ctx[row +  0 + l16] = f2bf(o0[r] * inv);
        ctx[row + 16 + l16] = f2bf(o1[r] * inv);
        ctx[row + 32 + l16] = f2bf(o2[r] * inv);
        ctx[row + 48 + l16] = f2bf(o3[r] * inv);
    }
}

// ===========================================================================
// Host-side orchestration
// ===========================================================================
extern "C" void kernel_launch(void* const* d_in, const int* in_sizes, int n_in,
                              void* d_out, int out_size, void* d_ws, size_t ws_size,
                              hipStream_t stream) {
    const float* q_in = (const float*)d_in[0];
    const float* k_in = (const float*)d_in[1];
    const float* v_in = (const float*)d_in[2];
    const float* w_q  = (const float*)d_in[3];
    const float* b_q  = (const float*)d_in[4];
    const float* w_k  = (const float*)d_in[5];
    const float* b_k  = (const float*)d_in[6];
    const float* w_v  = (const float*)d_in[7];
    const float* b_v  = (const float*)d_in[8];
    const float* w_o  = (const float*)d_in[9];
    const float* b_o  = (const float*)d_in[10];

    const size_t NTOK = (size_t)BB * SS;              // 4096 rows
    const size_t EACT = NTOK * DM;                    // 4,194,304 elements

    char* ws = (char*)d_ws;
    size_t off = 0;
    auto take = [&](size_t bytes) { char* p = ws + off; off += (bytes + 255) & ~(size_t)255; return p; };
    __bf16* qinB = (__bf16*)take(EACT * 2);
    __bf16* kinB = (__bf16*)take(EACT * 2);
    __bf16* vinB = (__bf16*)take(EACT * 2);
    __bf16* wqT  = (__bf16*)take((size_t)DM * DM * 2);
    __bf16* wkT  = (__bf16*)take((size_t)DM * KVD * 2);
    __bf16* wvT  = (__bf16*)take((size_t)DM * KVD * 2);
    __bf16* woT  = (__bf16*)take((size_t)DM * DM * 2);
    __bf16* Qb   = (__bf16*)take(NTOK * DM * 2);
    __bf16* Kb   = (__bf16*)take(NTOK * KVD * 2);
    __bf16* Vb   = (__bf16*)take(NTOK * KVD * 2);
    __bf16* VtB  = (__bf16*)take(NTOK * KVD * 2);
    __bf16* ctxB = (__bf16*)take(NTOK * DM * 2);
    (void)ws_size;

    // 1) activations f32 -> bf16
    {
        int n4 = (int)(EACT / 4);
        k_cvt4<<<n4 / 256, 256, 0, stream>>>((const float4*)q_in, (ushort4*)qinB, n4);
        k_cvt4<<<n4 / 256, 256, 0, stream>>>((const float4*)k_in, (ushort4*)kinB, n4);
        k_cvt4<<<n4 / 256, 256, 0, stream>>>((const float4*)v_in, (ushort4*)vinB, n4);
    }
    // 2) weights -> transposed bf16
    k_wtrans<<<(DM * DM) / 256, 256, 0, stream>>>(w_q, wqT, DM, DM);
    k_wtrans<<<(DM * KVD) / 256, 256, 0, stream>>>(w_k, wkT, DM, KVD);
    k_wtrans<<<(DM * KVD) / 256, 256, 0, stream>>>(w_v, wvT, DM, KVD);
    k_wtrans<<<(DM * DM) / 256, 256, 0, stream>>>(w_o, woT, DM, DM);

    // 3) projections (LDS-staged WMMA GEMMs), outputs bf16
    {
        int blkQ = (int)(NTOK / 128) * (DM / 64);     // 512
        int blkK = (int)(NTOK / 128) * (KVD / 64);    // 128
        k_gemm<true><<<blkQ, 256, 0, stream>>>(qinB, wqT, b_q, Qb, (int)NTOK, DM, DM);
        k_gemm<true><<<blkK, 256, 0, stream>>>(kinB, wkT, b_k, Kb, (int)NTOK, KVD, DM);
        k_gemm<true><<<blkK, 256, 0, stream>>>(vinB, wvT, b_v, Vb, (int)NTOK, KVD, DM);
    }

    // 4) V transpose for PV B-fragments
    k_vtrans<<<(int)(NTOK * KVD) / 256, 256, 0, stream>>>(Vb, VtB);

    // 5) attention (flash, WMMA)
    {
        int waves = BB * NH * (SS / 16);              // 4096
        k_attn<<<waves / 8, 256, 0, stream>>>(Qb, Kb, VtB, ctxB);
    }

    // 6) output projection -> d_out (f32)
    {
        int blk = (int)(NTOK / 128) * (DM / 64);      // 512
        k_gemm<false><<<blk, 256, 0, stream>>>(ctxB, woT, b_o, d_out, (int)NTOK, DM, DM);
    }
    (void)in_sizes; (void)n_in; (void)out_size;
}
